// Attention_34041910788382
// MI455X (gfx1250) — matrix operand order repaired and checked by reading, not compiled, not run on hardware
//
#include <hip/hip_runtime.h>
#include <hip/hip_bf16.h>

// Problem dims (fixed by the reference).
#define B_DIM 4096
#define T_DIM 16
#define H_DIM 1024
#define E_DIM 1024
#define K_DIM 1024   // both GEMMs have K = E = H = 1024 (compile-time)

typedef __bf16 bf16_t;
typedef __attribute__((ext_vector_type(16))) __bf16 v16bf;
typedef __attribute__((ext_vector_type(8)))  float   v8f;
typedef __attribute__((ext_vector_type(4)))  float   f32x4;

// ---------------------------------------------------------------------------
// WMMA fragment loaders: load fp32 from global, convert to bf16 in-register.
//
// A fragment (16x32, row-major source, lda = K):
//   lanes 0-15 : row M=lane,    elems 0-7 = K{0..7},   elems 8-15 = K{16..23}
//   lanes 16-31: row M=lane-16, elems 0-7 = K{8..15},  elems 8-15 = K{24..31}
// ---------------------------------------------------------------------------
__device__ __forceinline__ v16bf load_a_frag_f32(const float* __restrict__ row0,
                                                 long lda, int lane) {
    const float* p = row0 + (long)(lane & 15) * lda + ((lane >> 4) << 3);
    f32x4 x0 = *(const f32x4*)(p + 0);
    f32x4 x1 = *(const f32x4*)(p + 4);
    f32x4 x2 = *(const f32x4*)(p + 16);
    f32x4 x3 = *(const f32x4*)(p + 20);
    v16bf r;
#pragma unroll
    for (int i = 0; i < 4; ++i) {
        r[i + 0]  = (__bf16)x0[i];
        r[i + 4]  = (__bf16)x1[i];
        r[i + 8]  = (__bf16)x2[i];
        r[i + 12] = (__bf16)x3[i];
    }
    return r;
}

// B fragment (32x16). Weight stored N-major/K-contiguous (i.e. W[n*ldb + k]):
//   lanes 0-15 : column N=lane,     elems 0-7 = K{0..7},   elems 8-15 = K{16..23}
//   lanes 16-31: column N=lane-16,  elems 0-7 = K{8..15},  elems 8-15 = K{24..31}
__device__ __forceinline__ v16bf load_b_frag_f32(const float* __restrict__ col0,
                                                 long ldb, int lane) {
    const float* p = col0 + (long)(lane & 15) * ldb + ((lane >> 4) << 3);
    f32x4 x0 = *(const f32x4*)(p + 0);
    f32x4 x1 = *(const f32x4*)(p + 4);
    f32x4 x2 = *(const f32x4*)(p + 16);
    f32x4 x3 = *(const f32x4*)(p + 20);
    v16bf r;
#pragma unroll
    for (int i = 0; i < 4; ++i) {
        r[i + 0]  = (__bf16)x0[i];
        r[i + 4]  = (__bf16)x1[i];
        r[i + 8]  = (__bf16)x2[i];
        r[i + 12] = (__bf16)x3[i];
    }
    return r;
}

// ---------------------------------------------------------------------------
// GEMM: C[M,N] = A[M,K_DIM] (fp32, row-major) x W (fp32, N-major [N,K_DIM]).
// bf16 WMMA, fp32 accumulate. Block = 4 waves in 2x2, block tile 128x128,
// per-wave tile 64x64 (4x4 accumulators -> 16 WMMAs per 32-wide K-chunk,
// 32 b128 loads per chunk = 2 loads/WMMA). M%128==0, N%128==0 assumed.
// OUT_BF16 picks the destination (Cb bf16, else Cf fp32), row-major ldc=N.
// ---------------------------------------------------------------------------
template <bool OUT_BF16>
__global__ __launch_bounds__(128) void gemm_bf16wmma_kernel(
    const float* __restrict__ A, const float* __restrict__ W,
    float* __restrict__ Cf, bf16_t* __restrict__ Cb,
    int M, int N)
{
    const int lane = threadIdx.x & 31;
    const int wave = threadIdx.x >> 5;      // 0..3
    const int wm   = wave >> 1;             // 0..1
    const int wn   = wave & 1;              // 0..1
    const long m0 = (long)blockIdx.y * 128 + (long)wm * 64;
    const long n0 = (long)blockIdx.x * 128 + (long)wn * 64;
    const long lK = K_DIM, lN = N;

    v8f acc[4][4];
#pragma unroll
    for (int i = 0; i < 4; ++i)
#pragma unroll
        for (int j = 0; j < 4; ++j)
            acc[i][j] = (v8f){0.f,0.f,0.f,0.f,0.f,0.f,0.f,0.f};

#pragma unroll 2
    for (int k0 = 0; k0 < K_DIM; k0 += 32) {
        if (k0 + 32 < K_DIM) {  // compile-time-resolvable guard
            __builtin_prefetch(A + m0 * lK + k0 + 32, 0, 3);
            __builtin_prefetch(W + n0 * lK + k0 + 32, 0, 3);
        }
        v16bf af[4], bfr[4];
#pragma unroll
        for (int i = 0; i < 4; ++i)
            af[i] = load_a_frag_f32(A + (m0 + i * 16) * lK + k0, lK, lane);
#pragma unroll
        for (int j = 0; j < 4; ++j)
            bfr[j] = load_b_frag_f32(W + (n0 + j * 16) * lK + k0, lK, lane);
#pragma unroll
        for (int i = 0; i < 4; ++i)
#pragma unroll
            for (int j = 0; j < 4; ++j)
                acc[i][j] = __builtin_amdgcn_wmma_f32_16x16x32_bf16(
                    false, af[i], false, bfr[j], (short)0, acc[i][j], false, false);
    }

    // C/D layout: VGPR v, lanes 0-15 -> (M=v, N=lane); lanes 16-31 -> (M=v+8).
    const int col = lane & 15;
    const int r0  = (lane >> 4) << 3;
#pragma unroll
    for (int i = 0; i < 4; ++i)
#pragma unroll
        for (int j = 0; j < 4; ++j) {
            const long cbase = (m0 + i * 16 + r0) * lN + (n0 + j * 16 + col);
#pragma unroll
            for (int e = 0; e < 8; ++e) {
                if (OUT_BF16) Cb[cbase + (long)e * lN] = (bf16_t)acc[i][j][e];
                else          Cf[cbase + (long)e * lN] = acc[i][j][e];
            }
        }
}

// ---------------------------------------------------------------------------
// Per-step score: s[b,t] = (sum_h va[h] * tanh(q_t[b,h] + pk[b,t,h]) + va_b)
//                 * coverage[b,t].   One block (256 thr) per batch row.
// ---------------------------------------------------------------------------
__global__ __launch_bounds__(256) void score_kernel(
    const float* __restrict__ qn, const bf16_t* __restrict__ pk,
    const float* __restrict__ va_w, const float* __restrict__ va_b,
    const float* __restrict__ cov, float* __restrict__ s, int t)
{
    const int b = blockIdx.x;
    const int tid = threadIdx.x;
    const float*  qrow = qn + (size_t)b * H_DIM;
    const bf16_t* prow = pk + ((size_t)b * T_DIM + t) * H_DIM;

    float part = 0.f;
#pragma unroll
    for (int h = tid; h < H_DIM; h += 256)
        part += va_w[h] * tanhf(qrow[h] + (float)prow[h]);

    __shared__ float red[256];
    red[tid] = part;
    __syncthreads();
#pragma unroll
    for (int off = 128; off > 0; off >>= 1) {
        if (tid < off) red[tid] += red[tid + off];
        __syncthreads();
    }
    if (tid == 0)
        s[(size_t)b * T_DIM + t] = (red[0] + va_b[0]) * cov[(size_t)b * T_DIM + t];
}

// Softmax over T=16 per batch row; one thread per row.
__global__ __launch_bounds__(256) void softmax_kernel(
    const float* __restrict__ s, float* __restrict__ alphas)
{
    const int b = blockIdx.x * 256 + threadIdx.x;
    if (b >= B_DIM) return;
    float v[T_DIM];
    float mx = -3.4e38f;
#pragma unroll
    for (int t = 0; t < T_DIM; ++t) { v[t] = s[(size_t)b * T_DIM + t]; mx = fmaxf(mx, v[t]); }
    float sum = 0.f;
#pragma unroll
    for (int t = 0; t < T_DIM; ++t) { v[t] = __expf(v[t] - mx); sum += v[t]; }
    const float inv = 1.f / sum;
#pragma unroll
    for (int t = 0; t < T_DIM; ++t) alphas[(size_t)b * T_DIM + t] = v[t] * inv;
}

// Context: mt[b,e] = sum_t alphas[b,t] * topics[b,t,e]. One block per b,
// 256 threads x float4 = 1024 columns.
__global__ __launch_bounds__(256) void context_kernel(
    const float* __restrict__ alphas, const float* __restrict__ topics,
    float* __restrict__ mt)
{
    const int b = blockIdx.x;
    const int tid = threadIdx.x;
    __shared__ float al[T_DIM];
    if (tid < T_DIM) al[tid] = alphas[(size_t)b * T_DIM + tid];
    __syncthreads();

    const int e0 = tid * 4;
    f32x4 acc = (f32x4){0.f, 0.f, 0.f, 0.f};
#pragma unroll
    for (int t = 0; t < T_DIM; ++t) {
        const f32x4 tv = *(const f32x4*)(topics + ((size_t)b * T_DIM + t) * E_DIM + e0);
        const float a = al[t];
        acc.x += a * tv.x; acc.y += a * tv.y; acc.z += a * tv.z; acc.w += a * tv.w;
    }
    *(f32x4*)(mt + (size_t)b * E_DIM + e0) = acc;
}

// ---------------------------------------------------------------------------
extern "C" void kernel_launch(void* const* d_in, const int* in_sizes, int n_in,
                              void* d_out, int out_size, void* d_ws, size_t ws_size,
                              hipStream_t stream) {
    const float* query  = (const float*)d_in[0];  // [B,H]
    const float* topics = (const float*)d_in[1];  // [B,T,E]
    const float* cov    = (const float*)d_in[2];  // [B,T]
    const float* Ua_w   = (const float*)d_in[3];  // [H,E]  (N-major, K-contig: ready as WMMA B)
    const float* Wa_w   = (const float*)d_in[4];  // [H',H] (N-major, K-contig: ready as WMMA B)
    const float* va_w   = (const float*)d_in[5];  // [H]
    const float* va_b   = (const float*)d_in[6];  // [1]

    float* mt     = (float*)d_out;                         // [B,E]
    float* alphas = (float*)d_out + (size_t)B_DIM * E_DIM; // [B,T]

    // Workspace layout (~160.3 MB)
    char* ws = (char*)d_ws;
    size_t off = 0;
    bf16_t* pk = (bf16_t*)(ws + off); off += (size_t)B_DIM * T_DIM * H_DIM * sizeof(bf16_t);
    off = (off + 255) & ~(size_t)255;
    float* qA  = (float*)(ws + off);  off += (size_t)B_DIM * H_DIM * sizeof(float);
    float* qB  = (float*)(ws + off);  off += (size_t)B_DIM * H_DIM * sizeof(float);
    float* s   = (float*)(ws + off);  off += (size_t)B_DIM * T_DIM * sizeof(float);
    (void)ws_size; (void)in_sizes; (void)n_in; (void)out_size;

    // 1) proj_keys (bf16): pk[b*T+t, :] = topics[b,t,:] @ Ua_w^T
    {
        dim3 grid(H_DIM / 128, (B_DIM * T_DIM) / 128);
        gemm_bf16wmma_kernel<true><<<grid, 128, 0, stream>>>(
            topics, Ua_w, nullptr, pk, B_DIM * T_DIM, H_DIM);
    }

    // 2) Sequential coverage scan: q <- q @ Wa^T, then score for step t.
    {
        dim3 grid(H_DIM / 128, B_DIM / 128);
        const float* qcur = query;
        float* bufs[2] = {qA, qB};
        for (int t = 0; t < T_DIM; ++t) {
            float* qn = bufs[t & 1];
            gemm_bf16wmma_kernel<false><<<grid, 128, 0, stream>>>(
                qcur, Wa_w, qn, nullptr, B_DIM, H_DIM);
            score_kernel<<<B_DIM, 256, 0, stream>>>(qn, pk, va_w, va_b, cov, s, t);
            qcur = qn;
        }
    }

    // 3) Softmax over T
    softmax_kernel<<<B_DIM / 256, 256, 0, stream>>>(s, alphas);

    // 4) Context vector mt = alphas @ topics (per batch row)
    context_kernel<<<B_DIM, 256, 0, stream>>>(alphas, topics, mt);
}
